// LinearConv2d_34196529610881
// MI455X (gfx1250) — matrix operand order repaired and checked by reading, not compiled, not run on hardware
//
#include <hip/hip_runtime.h>
#include <hip/hip_bf16.h>

// ---- problem constants --------------------------------------------------
#define B_    32
#define CIN_  128
#define H_    56
#define W_    56
#define HW_   (H_ * W_)        // 3136
#define COUT_ 256
#define KK_   (CIN_ * 9)       // 1152
#define MPIX  (B_ * HW_)       // 100352

// ---- tiling -------------------------------------------------------------
#define BM 128                 // COUT per block   (GEMM M)
#define BN 128                 // pixels per block (GEMM N)
#define KC 32                  // K per chunk (bf16 WMMA K)
#define NCHUNK (KK_ / KC)      // 36 ; K reordered as (kh,kw,ci)
#define LDH 40                 // LDS row stride in halves (32 + 8 pad)

typedef __attribute__((ext_vector_type(16))) __bf16 v16bf;
typedef __attribute__((ext_vector_type(8)))  float  v8f;
typedef __attribute__((ext_vector_type(4)))  unsigned int u32x4;

union FragU { u32x4 q[2]; v16bf v; };
union PackU { __bf16 h[16]; u32x4 q[2]; };

__global__ __launch_bounds__(256)
void conv3x3_implicit_wmma(const float* __restrict__ X,   // [B][CIN][H][W]
                           const float* __restrict__ Wm,  // [COUT][1152]
                           const float* __restrict__ bias,// [COUT]
                           float* __restrict__ out)       // [B][COUT][H][W]
{
  __shared__ __align__(16) unsigned short As[2][BM][LDH]; // W tile [co][k]
  __shared__ __align__(16) unsigned short Bs[2][BN][LDH]; // x tile [pix][k]

  const int tid  = threadIdx.x;
  const int lane = tid & 31;
  const int l16  = lane & 15;
  const int lh   = lane >> 4;
  const int wid  = tid >> 5;
  const int wm   = wid & 3;            // co waves:  4 x 32
  const int wn   = wid >> 2;           // pix waves: 2 x 64

  const int pixBase = blockIdx.x * BN;
  const int coBase  = blockIdx.y * BM;

  // ---- staging maps (hoisted) ----
  const int xp  = tid & 127;                 // pix_sub
  const int xcb = (tid >> 7) * 16;           // ci_sub base (0|16)
  const int p   = pixBase + xp;
  const int pb  = p / HW_;
  const int phw = p - pb * HW_;
  const int ph  = phw / W_;
  const int pw  = phw - ph * W_;
  const float* xbase = X + (size_t)pb * (CIN_ * HW_) + (size_t)xcb * HW_;

  const int wc  = tid >> 1;                  // co_sub
  const int wkb = (tid & 1) * 16;            // ci_sub base (0|16)
  const float* wrow = Wm + (size_t)(coBase + wc) * KK_ + (size_t)wkb * 9;

  // pipelined staging registers
  float xv[16], wv[16], xmask;

  auto issue_loads = [&](int kc) {
    const int r  = kc >> 2;                  // kh*3+kw
    const int cb = (kc & 3) * 32;            // ci base
    const int dh = r / 3 - 1;
    const int dw = r - (r / 3) * 3 - 1;
    int h2 = ph + dh, w2 = pw + dw;
    const bool ok = ((unsigned)h2 < (unsigned)H_) & ((unsigned)w2 < (unsigned)W_);
    xmask = ok ? 1.f : 0.f;
    h2 = min(max(h2, 0), H_ - 1);            // clamp -> address always valid
    w2 = min(max(w2, 0), W_ - 1);
    const float* xs = xbase + (size_t)cb * HW_ + h2 * W_ + w2;
#pragma unroll
    for (int i = 0; i < 16; ++i) xv[i] = xs[(size_t)i * HW_];   // unconditional
    const float* ws = wrow + (size_t)cb * 9 + r;
#pragma unroll
    for (int i = 0; i < 16; ++i) wv[i] = ws[(size_t)i * 9];
  };

  v8f acc[2][4];
#pragma unroll
  for (int mt = 0; mt < 2; ++mt)
#pragma unroll
    for (int nt = 0; nt < 4; ++nt)
      acc[mt][nt] = (v8f){0.f, 0.f, 0.f, 0.f, 0.f, 0.f, 0.f, 0.f};

  issue_loads(0);   // prologue

  for (int kc = 0; kc < NCHUNK; ++kc) {
    const int buf = kc & 1;

    // ---- convert + store staged regs into LDS[buf] ----
    {
      PackU px;
#pragma unroll
      for (int i = 0; i < 16; ++i) px.h[i] = (__bf16)(xv[i] * xmask);
      u32x4* dx = (u32x4*)&Bs[buf][xp][xcb];
      dx[0] = px.q[0];
      dx[1] = px.q[1];

      PackU pk;
#pragma unroll
      for (int i = 0; i < 16; ++i) pk.h[i] = (__bf16)wv[i];
      u32x4* dw_ = (u32x4*)&As[buf][wc][wkb];
      dw_[0] = pk.q[0];
      dw_[1] = pk.q[1];
    }

    __syncthreads();   // single barrier per chunk (double buffered)

    // ---- issue next chunk's global loads; latency hides under WMMAs ----
    const int kn = (kc + 1 == NCHUNK) ? 0 : kc + 1;
    issue_loads(kn);

    // ---- fragments per ISA layouts ----
    FragU a[2];
#pragma unroll
    for (int mt = 0; mt < 2; ++mt) {
      const unsigned short* s = &As[buf][wm * 32 + mt * 16 + l16][lh * 8];
      a[mt].q[0] = *(const u32x4*)s;
      a[mt].q[1] = *(const u32x4*)(s + 16);
    }
    FragU bfr[4];
#pragma unroll
    for (int nt = 0; nt < 4; ++nt) {
      const unsigned short* s = &Bs[buf][wn * 64 + nt * 16 + l16][lh * 8];
      bfr[nt].q[0] = *(const u32x4*)s;
      bfr[nt].q[1] = *(const u32x4*)(s + 16);
    }

#pragma unroll
    for (int mt = 0; mt < 2; ++mt)
#pragma unroll
      for (int nt = 0; nt < 4; ++nt)
        acc[mt][nt] = __builtin_amdgcn_wmma_f32_16x16x32_bf16(
            false, a[mt].v, false, bfr[nt].v, (short)0, acc[mt][nt],
            false, false);
  }

  // ---- epilogue: D VGPR rr -> (co = base + rr + 8*lh, pix = base + l16) ----
#pragma unroll
  for (int nt = 0; nt < 4; ++nt) {
    const int pix = pixBase + wn * 64 + nt * 16 + l16;
    const int ob  = pix / HW_;
    const int ohw = pix - ob * HW_;
    float* op = out + (size_t)ob * (COUT_ * HW_) + ohw;
#pragma unroll
    for (int mt = 0; mt < 2; ++mt) {
      const int co0 = coBase + wm * 32 + mt * 16 + lh * 8;
#pragma unroll
      for (int rr = 0; rr < 8; ++rr) {
        const int co = co0 + rr;
        op[(size_t)co * HW_] = acc[mt][nt][rr] + bias[co];
      }
    }
  }
}

extern "C" void kernel_launch(void* const* d_in, const int* in_sizes, int n_in,
                              void* d_out, int out_size, void* d_ws, size_t ws_size,
                              hipStream_t stream) {
  const float* x  = (const float*)d_in[0];
  const float* W  = (const float*)d_in[1];
  const float* b  = (const float*)d_in[2];
  float* out = (float*)d_out;
  dim3 grid(MPIX / BN, COUT_ / BM);   // (784, 2)
  conv3x3_implicit_wmma<<<grid, dim3(256), 0, stream>>>(x, W, b, out);
}
